// DynamicRNNEncoder_9689446220126
// MI455X (gfx1250) — compile-verified, matrix operand order checked by reading, not run on hardware
//
#include <hip/hip_runtime.h>
#include <hip/hip_bf16.h>

// ---------------------------------------------------------------------------
// DynamicRNNEncoder on MI455X (gfx1250, wave32, WMMA bf16).
//
// 4 groups (16 batch rows) x 16 WGs: hidden dim split so each WG's main-LSTM
// weight slice (256 KB bf16) is LDS-resident all 256 steps (staged once via
// TDM tensor_load_to_lds with HW pad -> bank-conflict-free 520-elem rows).
// Dyn weights stream from L2 on dedicated waves. Per-step activation panels
// staged with async global->LDS (ASYNCcnt). Group sync: s_cluster_barrier +
// global atomic counter fallback with s_sleep polling.
// ---------------------------------------------------------------------------

typedef __attribute__((ext_vector_type(16))) __bf16 v16bf;
typedef __attribute__((ext_vector_type(8)))  float  v8f;
typedef __attribute__((ext_vector_type(4)))  unsigned int v4u;
typedef __attribute__((ext_vector_type(8)))  int v8i;
typedef __attribute__((ext_vector_type(4)))  int v4i;

#define BB   64
#define TT   256
#define HH   512
#define DD   256
#define ENN  256
#define ETT  256
#define VDD  900          // dynamic vocab slots (1000 - 100 static)
#define NGRP 4            // batch groups of 16 rows
#define NWG  16           // workgroups per group (cluster-sized)
#define THR  256          // 8 waves: 4 main-gate + 4 dyn-gate

// LDS row strides padded +8 elements (16B): 16 lanes on consecutive rows land
// 4 banks apart -> conflict-free b128 reads.
#define WPAD 520
#define APAD 776
#define HPAD 264

// LDS layout (bytes); total 323,392 <= 327,680 (320 KB). Dynamic-LDS base
// assumed at LDS offset 0 (no static __shared__ in this kernel).
#define OFF_WIH  0
#define OFF_WHH  (OFF_WIH + 128 * WPAD * 2)
#define OFF_A    (OFF_WHH + 128 * WPAD * 2)
#define OFF_HT   (OFF_A   + 16 * APAD * 2)
#define OFF_HD   (OFF_HT  + 16 * HPAD * 2)
#define OFF_GM   (OFF_HD  + 16 * HPAD * 2)
#define OFF_GD   (OFF_GM  + 16 * 128 * 4)
#define OFF_C    (OFF_GD  + 16 * 64 * 4)
#define OFF_BM   (OFF_C   + 16 * 32 * 4)
#define OFF_BD   (OFF_BM  + 128 * 4)
#define OFF_META (OFF_BD  + 64 * 4)
#define SMEM_BYTES (OFF_META + 5 * 16 * 4)

union FragU { v16bf v; uint4 q[2]; };

__device__ __forceinline__ v16bf ld_frag(const __bf16* p, int k0) {
  FragU f;
  f.q[0] = *(const uint4*)(p + k0);
  f.q[1] = *(const uint4*)(p + k0 + 16);
  return f.v;
}

__device__ __forceinline__ float sigm_f(float x) { return 1.f / (1.f + __expf(-x)); }
__device__ __forceinline__ float tanh_f(float x) {
  x = fminf(fmaxf(x, -15.f), 15.f);
  float e = __expf(2.f * x);
  return (e - 1.f) / (e + 1.f);
}

#define WMMA_BF16(a, b, c) \
  __builtin_amdgcn_wmma_f32_16x16x32_bf16(false, (a), false, (b), (short)0, (c), false, false)

// ---- async global->LDS copy (GLOBAL_LOAD_ASYNC_TO_LDS_B128, ASYNCcnt) ----
__device__ __forceinline__ void async_g2l_b128(unsigned lds_byte_addr, const void* g) {
  asm volatile("global_load_async_to_lds_b128 %0, %1, off"
               :: "v"(lds_byte_addr), "v"(g) : "memory");
}
__device__ __forceinline__ void async_wait0() {
  asm volatile("s_wait_asynccnt 0x0" ::: "memory");
}

// ---- TDM: 2D tile (rows x 512 bf16) global->LDS with HW padding to 520 ----
// pad_interval=7 (256 DWORDs = one 1024B row), pad_amount=3 (4 DWORDs = 16B).
#if __has_builtin(__builtin_amdgcn_tensor_load_to_lds)
#define HAVE_TDM 1
__device__ __forceinline__ void tdm_load_rows(unsigned lds_addr, const __bf16* gsrc,
                                              int rows) {
  unsigned long long ga = (unsigned long long)(const void*)gsrc;
  v4u g0;
  g0.x = 1u;                                              // count=1 (valid), user mode
  g0.y = lds_addr;                                        // LDS byte address
  g0.z = (unsigned)(ga & 0xffffffffu);                    // global_addr[95:64]
  g0.w = (unsigned)((ga >> 32) & 0x01ffffffull) | (2u << 30);  // addr hi | type=2
  v8i g1;
  g1[0] = (1 << 16) | (1 << 20) | (7 << 22) | (3 << 25);  // data_size=2B, pad_en, interval, amount
  g1[1] = (512 & 0xffff) << 16;                           // tensor_dim0 = 512 (lo16)
  g1[2] = ((512 >> 16) & 0xffff) | ((rows & 0xffff) << 16); // dim0 hi | tensor_dim1 lo
  g1[3] = ((rows >> 16) & 0xffff) | ((512 & 0xffff) << 16); // dim1 hi | tile_dim0 = 512
  g1[4] = (rows & 0xffff);                                // tile_dim1 = rows, tile_dim2 = 0
  g1[5] = 512;                                            // tensor_dim0_stride = 512
  g1[6] = 0;
  g1[7] = 0;
  v4i z4 = {0, 0, 0, 0};
#if defined(__clang_major__) && __clang_major__ >= 23
  v8i z8 = {0, 0, 0, 0, 0, 0, 0, 0};
  __builtin_amdgcn_tensor_load_to_lds(g0, g1, z4, z4, z8, 0);
#else
  __builtin_amdgcn_tensor_load_to_lds(g0, g1, z4, z4, 0);
#endif
}
#else
#define HAVE_TDM 0
#endif

// ---------------------------------------------------------------------------
__global__ void prep_kernel(const float* Wih, const float* Whh,
                            const float* Wdih, const float* Wdhh,
                            const float* eN, const float* eT, const float* h0,
                            __bf16* Wihb, __bf16* Whhb, __bf16* Wdihb, __bf16* Wdhhb,
                            __bf16* eNb, __bf16* eTb, __bf16* hbuf,
                            unsigned char* seen, unsigned int* ctr) {
  const size_t NW = (size_t)4 * HH * HH;  // 1,048,576 (largest region)
  for (size_t i = (size_t)blockIdx.x * blockDim.x + threadIdx.x; i < NW;
       i += (size_t)gridDim.x * blockDim.x) {
    Wihb[i] = (__bf16)Wih[i];
    Whhb[i] = (__bf16)Whh[i];
    if (i < (size_t)4 * DD * (ENN + HH)) Wdihb[i] = (__bf16)Wdih[i];
    if (i < (size_t)4 * DD * DD)         Wdhhb[i] = (__bf16)Wdhh[i];
    if (i < (size_t)300 * ENN)           eNb[i]  = (__bf16)eN[i];
    if (i < (size_t)100 * ETT)           eTb[i]  = (__bf16)eT[i];
    if (i < (size_t)BB * VDD)            seen[i] = 0;
    if (i < (size_t)NGRP * 16 * HH)      hbuf[i] = (__bf16)h0[i & (HH - 1)];
    if (i < 64)                          ctr[i]  = 0;
  }
}

// ---------------------------------------------------------------------------
__global__ __launch_bounds__(THR, 1)
void rnn_kernel(const int* n_input, const int* t_input,
                const int* static2dense, const int* dynamic2dense,
                const float* b_ih, const float* b_hh,
                const float* bd_ih, const float* bd_hh,
                const float* cell_init,
                const float* dyn_init_h, const float* dyn_init_c,
                const __bf16* Wihb, const __bf16* Whhb,
                const __bf16* Wdihb, const __bf16* Wdhhb,
                const __bf16* embNb, const __bf16* embTb,
                __bf16* hbufAll, unsigned char* seenArr, unsigned int* ctrAll,
                __bf16* Hd, __bf16* Cd, float* out) {
  extern __shared__ unsigned char smem[];
  __bf16 (*sWih)[WPAD] = (__bf16(*)[WPAD])(smem + OFF_WIH);  // [128][520] gate-major rows
  __bf16 (*sWhh)[WPAD] = (__bf16(*)[WPAD])(smem + OFF_WHH);
  __bf16 (*sA)[APAD]   = (__bf16(*)[APAD])(smem + OFF_A);    // [16][776]: [n_emb|h]
  __bf16 (*sHT)[HPAD]  = (__bf16(*)[HPAD])(smem + OFF_HT);   // h_tensor
  __bf16 (*sHD)[HPAD]  = (__bf16(*)[HPAD])(smem + OFF_HD);   // hd_prev
  float (*sGM)[128]    = (float(*)[128])(smem + OFF_GM);
  float (*sGD)[64]     = (float(*)[64])(smem + OFF_GD);
  float (*sC)[32]      = (float(*)[32])(smem + OFF_C);
  float* sBiasM        = (float*)(smem + OFF_BM);
  float* sBiasD        = (float*)(smem + OFF_BD);
  int* sNT   = (int*)(smem + OFF_META);
  int* sS2D  = sNT + 16;
  int* sDidx = sNT + 32;
  int* sStat = sNT + 48;
  int* sSeen = sNT + 64;

  const int tid  = threadIdx.x;
  const int lane = tid & 31;
  const int wid  = tid >> 5;
  const int n    = lane & 15;
  const int hi   = lane >> 4;
  const int g    = blockIdx.x >> 4;
  const int w    = blockIdx.x & 15;
  const int rbase = g * 16;
  __bf16* hbuf = hbufAll + (size_t)g * 16 * HH;
  unsigned int* ctr = ctrAll + g * 16;

  // ---- one-time: stage main weight slice (256 KB) into LDS ----
#if HAVE_TDM
  if (wid == 0) {
#pragma unroll
    for (int G4 = 0; G4 < 4; ++G4) {
      tdm_load_rows((unsigned)(OFF_WIH + G4 * 32 * WPAD * 2),
                    Wihb + (size_t)(G4 * HH + 32 * w) * HH, 32);
      tdm_load_rows((unsigned)(OFF_WHH + G4 * 32 * WPAD * 2),
                    Whhb + (size_t)(G4 * HH + 32 * w) * HH, 32);
    }
    __builtin_amdgcn_s_wait_tensorcnt(0);
  }
#else
  for (int e = tid * 8; e < 128 * 512; e += THR * 8) {
    int r = e >> 9, c = e & 511;
    size_t gr = (size_t)((r >> 5) * HH + 32 * w + (r & 31));
    *(uint4*)&sWih[r][c] = *(const uint4*)&Wihb[gr * HH + c];
    *(uint4*)&sWhh[r][c] = *(const uint4*)&Whhb[gr * HH + c];
  }
#endif
  if (tid < 128) sBiasM[tid] = b_ih[(tid >> 5) * HH + 32 * w + (tid & 31)]
                             + b_hh[(tid >> 5) * HH + 32 * w + (tid & 31)];
  if (tid < 64)  sBiasD[tid] = bd_ih[(tid >> 4) * DD + 16 * w + (tid & 15)]
                             + bd_hh[(tid >> 4) * DD + 16 * w + (tid & 15)];
#pragma unroll
  for (int q = 0; q < 2; ++q) {
    int idx = tid * 2 + q;
    sC[idx >> 5][idx & 31] = cell_init[32 * w + (idx & 31)];
  }

  // ---- loop-invariant per-lane fragment pointers ----
  const int G = wid & 3;
  const __bf16* pW0ih = &sWih[G * 32 + n][0] + 8 * hi;
  const __bf16* pW1ih = &sWih[G * 32 + 16 + n][0] + 8 * hi;
  const __bf16* pW0hh = &sWhh[G * 32 + n][0] + 8 * hi;
  const __bf16* pW1hh = &sWhh[G * 32 + 16 + n][0] + 8 * hi;
  const __bf16* pDI = Wdihb + (size_t)(G * DD + 16 * w + n) * (ENN + HH) + 8 * hi;
  const __bf16* pDH = Wdhhb + (size_t)(G * DD + 16 * w + n) * DD + 8 * hi;
  const __bf16* pAne = &sA[n][0] + 8 * hi;
  const __bf16* pAh  = pAne + ENN;
  const __bf16* pHT  = &sHT[n][0] + 8 * hi;
  const __bf16* pHD  = &sHD[n][0] + 8 * hi;

  __syncthreads();

  for (int t = 0; t < TT; ++t) {
    // ===================== stage phase (state of step t-1) ==================
    if (tid < 16) {
      int r  = rbase + tid;
      int nt = n_input[(size_t)r * TT + t];
      int tk = t_input[(size_t)r * TT + t];
      int s2 = static2dense[tk];
      int di = dynamic2dense[tk];
      sNT[tid]   = nt;
      sS2D[tid]  = s2;
      sDidx[tid] = di;
      sStat[tid] = (s2 != 0) || (tk == 0);   // static ids are exactly range(100)
      sSeen[tid] = (int)seenArr[(size_t)r * VDD + di];
    }
    __syncthreads();
    {
      int m  = tid >> 4;
      unsigned ldsA = (unsigned)(OFF_A + m * APAD * 2);
      // h panel: 16x512 bf16, 4 async b128 per thread
      int ch = (tid & 15) * 32;
      const __bf16* hsrc = hbuf + (size_t)m * HH + ch;
#pragma unroll
      for (int q = 0; q < 4; ++q)
        async_g2l_b128(ldsA + (unsigned)((ENN + ch + q * 8) * 2), hsrc + q * 8);
      // n_emb panel: 16x256, 2 async b128 per thread
      int c2 = (tid & 15) * 16;
      const __bf16* ne = embNb + (size_t)sNT[m] * ENN + c2;
      async_g2l_b128(ldsA + (unsigned)(c2 * 2), ne);
      async_g2l_b128(ldsA + (unsigned)(c2 * 2 + 16), ne + 8);
      // hd_prev + h_tensor panels
      unsigned ldsHD = (unsigned)(OFF_HD + m * HPAD * 2 + c2 * 2);
      unsigned ldsHT = (unsigned)(OFF_HT + m * HPAD * 2 + c2 * 2);
      if (sSeen[m]) {
        const __bf16* ph = Hd + ((size_t)(rbase + m) * VDD + sDidx[m]) * DD + c2;
        async_g2l_b128(ldsHD, ph);
        async_g2l_b128(ldsHD + 16, ph + 8);
        if (!sStat[m]) {
          async_g2l_b128(ldsHT, ph);
          async_g2l_b128(ldsHT + 16, ph + 8);
        }
      } else {
        __bf16 tmp[16];
#pragma unroll
        for (int q = 0; q < 16; ++q) tmp[q] = (__bf16)dyn_init_h[c2 + q];
        *(uint4*)&sHD[m][c2]     = *(uint4*)&tmp[0];
        *(uint4*)&sHD[m][c2 + 8] = *(uint4*)&tmp[8];
        if (!sStat[m]) {
          *(uint4*)&sHT[m][c2]     = *(uint4*)&tmp[0];
          *(uint4*)&sHT[m][c2 + 8] = *(uint4*)&tmp[8];
        }
      }
      if (sStat[m]) {
        const __bf16* pt = embTb + (size_t)sS2D[m] * ETT + c2;
        async_g2l_b128(ldsHT, pt);
        async_g2l_b128(ldsHT + 16, pt + 8);
      }
      async_wait0();
    }
    __syncthreads();

    // ===================== WMMA phase ======================================
    if (wid < 4) {
      v8f a0, a1;
      {
        float b0 = sBiasM[G * 32 + n], b1 = sBiasM[G * 32 + 16 + n];
#pragma unroll
        for (int v = 0; v < 8; ++v) { a0[v] = b0; a1[v] = b1; }
      }
#pragma unroll
      for (int ck = 0; ck < 16; ++ck) {   // ih: x = [n_emb | h_tensor]
        v16bf a = (ck < 8) ? ld_frag(pAne, ck * 32) : ld_frag(pHT, ck * 32 - 256);
        a0 = WMMA_BF16(a, ld_frag(pW0ih, ck * 32), a0);
        a1 = WMMA_BF16(a, ld_frag(pW1ih, ck * 32), a1);
      }
#pragma unroll
      for (int ck = 0; ck < 16; ++ck) {   // hh: A = h
        v16bf a = ld_frag(pAh, ck * 32);
        a0 = WMMA_BF16(a, ld_frag(pW0hh, ck * 32), a0);
        a1 = WMMA_BF16(a, ld_frag(pW1hh, ck * 32), a1);
      }
#pragma unroll
      for (int v = 0; v < 8; ++v) {
        sGM[v + 8 * hi][G * 32 + n]      = a0[v];
        sGM[v + 8 * hi][G * 32 + 16 + n] = a1[v];
      }
    } else {
      v8f d0;
      {
        float b0 = sBiasD[G * 16 + n];
#pragma unroll
        for (int v = 0; v < 8; ++v) d0[v] = b0;
      }
#pragma unroll
      for (int ck = 0; ck < 24; ++ck)     // ih: A = [n_emb | h], K = 768
        d0 = WMMA_BF16(ld_frag(pAne, ck * 32), ld_frag(pDI, ck * 32), d0);
#pragma unroll
      for (int ck = 0; ck < 8; ++ck)      // hh: A = hd_prev, K = 256
        d0 = WMMA_BF16(ld_frag(pHD, ck * 32), ld_frag(pDH, ck * 32), d0);
#pragma unroll
      for (int v = 0; v < 8; ++v) sGD[v + 8 * hi][G * 16 + n] = d0[v];
    }
    __syncthreads();

    // ===================== epilogue ========================================
#pragma unroll
    for (int q = 0; q < 2; ++q) {
      int idx = tid * 2 + q;
      int m = idx >> 5, u = idx & 31;
      float i_ = sigm_f(sGM[m][u]);
      float f_ = sigm_f(sGM[m][32 + u]);
      float g_ = tanh_f(sGM[m][64 + u]);
      float o_ = sigm_f(sGM[m][96 + u]);
      float cn = f_ * sC[m][u] + i_ * g_;
      sC[m][u] = cn;
      float hn = o_ * tanh_f(cn);
      int jcol = 32 * w + u;
      hbuf[(size_t)m * HH + jcol] = (__bf16)hn;
      out[((size_t)(rbase + m) * TT + t) * HH + jcol] = hn;
    }
    {
      int m = tid >> 4, ud = tid & 15;
      if (!sStat[m]) {
        int U = 16 * w + ud;
        size_t slot = ((size_t)(rbase + m) * VDD + sDidx[m]) * DD + U;
        float cdp = sSeen[m] ? (float)Cd[slot] : dyn_init_c[U];
        float i_ = sigm_f(sGD[m][ud]);
        float f_ = sigm_f(sGD[m][16 + ud]);
        float g_ = tanh_f(sGD[m][32 + ud]);
        float o_ = sigm_f(sGD[m][48 + ud]);
        float cdn = f_ * cdp + i_ * g_;
        Hd[slot] = (__bf16)(o_ * tanh_f(cdn));
        Cd[slot] = (__bf16)cdn;
      }
    }
    if (w == 0 && tid < 16 && !sStat[tid])
      seenArr[(size_t)(rbase + tid) * VDD + sDidx[tid]] = 1;

    // ===================== group-wide step barrier =========================
    __threadfence();
    __syncthreads();
    if (tid == 0)
      __hip_atomic_fetch_add(ctr, 1u, __ATOMIC_RELEASE, __HIP_MEMORY_SCOPE_AGENT);
    __builtin_amdgcn_s_cluster_barrier();  // HW fast path when dispatched as a cluster
    if (wid == 0) {
      unsigned int target = (unsigned int)(NWG * (t + 1));
      while (__hip_atomic_load(ctr, __ATOMIC_ACQUIRE, __HIP_MEMORY_SCOPE_AGENT) < target)
        __builtin_amdgcn_s_sleep(2);
    }
    __syncthreads();
  }
}

// ---------------------------------------------------------------------------
extern "C" void kernel_launch(void* const* d_in, const int* in_sizes, int n_in,
                              void* d_out, int out_size, void* d_ws, size_t ws_size,
                              hipStream_t stream) {
  const int*   n_input       = (const int*)d_in[0];
  const int*   t_input       = (const int*)d_in[1];
  // d_in[2] = is_static_vocab (derivable from static2dense; unused)
  const int*   static2dense  = (const int*)d_in[3];
  const int*   dynamic2dense = (const int*)d_in[4];
  const float* embN          = (const float*)d_in[5];
  const float* embT          = (const float*)d_in[6];
  const float* W_ih          = (const float*)d_in[7];
  const float* W_hh          = (const float*)d_in[8];
  const float* b_ih          = (const float*)d_in[9];
  const float* b_hh          = (const float*)d_in[10];
  const float* Wd_ih         = (const float*)d_in[11];
  const float* Wd_hh         = (const float*)d_in[12];
  const float* bd_ih         = (const float*)d_in[13];
  const float* bd_hh         = (const float*)d_in[14];
  const float* hid_init      = (const float*)d_in[15];
  const float* cell_init     = (const float*)d_in[16];
  const float* dyn_init_h    = (const float*)d_in[17];
  const float* dyn_init_c    = (const float*)d_in[18];
  float* out = (float*)d_out;

  unsigned char* p = (unsigned char*)d_ws;
  size_t off = 0;
  __bf16* Wihb  = (__bf16*)(p + off); off += (size_t)4 * HH * HH * 2;
  __bf16* Whhb  = (__bf16*)(p + off); off += (size_t)4 * HH * HH * 2;
  __bf16* Wdihb = (__bf16*)(p + off); off += (size_t)4 * DD * (ENN + HH) * 2;
  __bf16* Wdhhb = (__bf16*)(p + off); off += (size_t)4 * DD * DD * 2;
  __bf16* eNb   = (__bf16*)(p + off); off += (size_t)300 * ENN * 2;
  __bf16* eTb   = (__bf16*)(p + off); off += (size_t)100 * ETT * 2;
  unsigned char* seen = (unsigned char*)(p + off);
  off = (off + (size_t)BB * VDD + 255) & ~(size_t)255;
  __bf16* hbuf = (__bf16*)(p + off); off += (size_t)NGRP * 16 * HH * 2;
  unsigned int* ctr = (unsigned int*)(p + off); off += 256;
  __bf16* Hd = (__bf16*)(p + off); off += (size_t)BB * VDD * DD * 2;
  __bf16* Cd = (__bf16*)(p + off); off += (size_t)BB * VDD * DD * 2;
  (void)ws_size; (void)in_sizes; (void)n_in; (void)out_size;

  (void)hipFuncSetAttribute((const void*)rnn_kernel,
                            hipFuncAttributeMaxDynamicSharedMemorySize,
                            (int)SMEM_BYTES);

  prep_kernel<<<2048, 512, 0, stream>>>(W_ih, W_hh, Wd_ih, Wd_hh, embN, embT, hid_init,
                                        Wihb, Whhb, Wdihb, Wdhhb, eNb, eTb, hbuf,
                                        seen, ctr);

  rnn_kernel<<<NGRP * NWG, THR, SMEM_BYTES, stream>>>(
      n_input, t_input, static2dense, dynamic2dense,
      b_ih, b_hh, bd_ih, bd_hh, cell_init, dyn_init_h, dyn_init_c,
      Wihb, Whhb, Wdihb, Wdhhb, eNb, eTb,
      hbuf, seen, ctr, Hd, Cd, out);
}
